// Tokenizer_66614942761435
// MI455X (gfx1250) — matrix operand order repaired and checked by reading, not compiled, not run on hardware
//
#include <hip/hip_runtime.h>
#include <hip/hip_bf16.h>

// ---------------------------------------------------------------------------
// Tokenizer: concat(2104) -> ReLU(X@W1+b1) -> (@W2+b2) -> mask -> mean over S.
//   out = ( (sum_s mask_s * h[s,:]) @ W2 + cnt*b2 ) / 50   (50x cut on GEMM2)
// GEMM1 (453 GFLOP) dominates -> bf16 WMMA, f32 acc, compute-bound.
// W1/W2 pre-packed in WMMA-B fragment order so LDS tiles are linear copies
// staged with gfx1250 async global->LDS DMA, double-buffered + pipelined.
// B fragments double-buffered in registers to overlap ds_load with WMMA.
// ---------------------------------------------------------------------------

#define BN_TOT 1024   // B*N
#define S_DIM  50
#define K1     2104   // true feature dim
#define KP     2112   // padded feature dim = 66 * 32
#define MP     64     // padded rows per (b,n)
#define TOK    1024
#define KSTEPS (KP / 32)   // 66
#define NTILE1 352         // 2112 = 6 * 352
#define NSUB1  11          // per-wave 16-col subtiles in gemm1
#define NSUB2  4           // per-wave 16-col subtiles in gemm2
#define ASTRIDE 20         // padded A row stride in dwords (bank spread)

#if __has_builtin(__builtin_amdgcn_global_load_async_to_lds_b128) && \
    __has_builtin(__builtin_amdgcn_s_wait_asynccnt)
#define USE_ASYNC_LDS 1
#else
#define USE_ASYNC_LDS 0
#endif

typedef __attribute__((ext_vector_type(16))) __bf16 v16bf;
typedef __attribute__((ext_vector_type(8)))  float  v8f;
typedef __attribute__((ext_vector_type(4)))  int    v4i;

typedef __attribute__((address_space(1))) v4i* gv4i_ptr;   // global int4*
typedef __attribute__((address_space(3))) v4i* lv4i_ptr;   // LDS int4*

union Frag { v16bf v; uint4 q[2]; };

__device__ __forceinline__ unsigned short f2bf(float f) {
  unsigned u = __float_as_uint(f);
  u += 0x7FFFu + ((u >> 16) & 1u);   // round-to-nearest-even
  return (unsigned short)(u >> 16);
}

// 16-byte global -> LDS copy (async DMA on CDNA5 when available)
__device__ __forceinline__ void cp16(void* lds_dst, const void* g_src) {
#if USE_ASYNC_LDS
  __builtin_amdgcn_global_load_async_to_lds_b128(
      (gv4i_ptr)g_src, (lv4i_ptr)lds_dst, 0, 0);
#else
  *(uint4*)lds_dst = *(const uint4*)g_src;
#endif
}

__device__ __forceinline__ void cp_wait() {
#if USE_ASYNC_LDS
  __builtin_amdgcn_s_wait_asynccnt(0);
#endif
}

// ---------------------------------------------------------------------------
// Pack concat features -> bf16, zero-padded to [BN][64][2112]
// ---------------------------------------------------------------------------
__global__ void pack_x_kernel(const float* __restrict__ emb,
                              const float* __restrict__ vis,
                              const float* __restrict__ bbox,
                              const float* __restrict__ kp,
                              unsigned short* __restrict__ Xb) {
  long long idx = (long long)blockIdx.x * blockDim.x + threadIdx.x;
  const long long total = (long long)BN_TOT * MP * KP;
  if (idx >= total) return;
  int c  = (int)(idx % KP);
  int r  = (int)((idx / KP) % MP);
  int bn = (int)(idx / ((long long)KP * MP));
  float val = 0.f;
  if (r < S_DIM && c < K1) {
    long long row = (long long)bn * S_DIM + r;
    if (c < 2048)       val = emb[row * 2048 + c];
    else if (c == 2048) val = vis[row];
    else if (c < 2053)  val = bbox[row * 4 + (c - 2049)];
    else                val = kp[row * 51 + (c - 2053)];
  }
  Xb[idx] = f2bf(val);
}

// Pack weights into WMMA-B fragment order:
//   Wp[((ks*cols_pad + n)*16 + p)] = { bf16 W[32ks+2p][n], bf16 W[32ks+2p+1][n] }
__global__ void pack_wp_kernel(const float* __restrict__ W,
                               unsigned* __restrict__ Wp,
                               int rows_in, int cols_in, int cols_pad) {
  long long idx = (long long)blockIdx.x * blockDim.x + threadIdx.x;
  const long long total = (long long)KSTEPS * cols_pad * 16;
  if (idx >= total) return;
  int p  = (int)(idx & 15);
  int n  = (int)((idx >> 4) % cols_pad);
  int ks = (int)(idx / (16LL * cols_pad));
  int k0 = ks * 32 + 2 * p;
  float f0 = (k0 < rows_in && n < cols_in) ? W[(long long)k0 * cols_in + n] : 0.f;
  float f1 = (k0 + 1 < rows_in && n < cols_in) ? W[(long long)(k0 + 1) * cols_in + n] : 0.f;
  Wp[idx] = (unsigned)f2bf(f0) | ((unsigned)f2bf(f1) << 16);
}

// Zero hbar accumulator + compute per-(b,n) mask counts.
__global__ void init_kernel(float* __restrict__ hbar,
                            float* __restrict__ cnt,
                            const int* __restrict__ mask) {
  long long idx = (long long)blockIdx.x * blockDim.x + threadIdx.x;
  const long long nh = (long long)BN_TOT * KP;
  if (idx < nh) hbar[idx] = 0.f;
  if (idx < BN_TOT) {
    float c = 0.f;
    for (int s = 0; s < S_DIM; ++s)
      c += (mask[idx * S_DIM + s] != 0) ? 1.f : 0.f;
    cnt[idx] = c;
  }
}

// ---------------------------------------------------------------------------
// GEMM1: per (b,n) block: h = relu(X@W1+b1); hbar += mask-weighted rowsum.
// 8 waves = 4(M) x 2(N); Ntile = 352 (6 exact passes); double-buffered LDS,
// pipelined (async) global->LDS staging, register-double-buffered B frags.
// ---------------------------------------------------------------------------
__global__ void __launch_bounds__(256)
gemm1_kernel(const unsigned short* __restrict__ Xb,
             const unsigned* __restrict__ W1p,
             const float* __restrict__ b1,
             const int* __restrict__ mask,
             float* __restrict__ hbar) {
  __shared__ unsigned AsBuf[2][MP * ASTRIDE];   // 2 x 5.0 KB
  __shared__ unsigned BsBuf[2][NTILE1 * 16];    // 2 x 22.0 KB (fragment-major)
  __shared__ unsigned char smask[MP];

  const int tid  = threadIdx.x;
  const int bn   = blockIdx.x;
  const int lane = tid & 31;
  const int l15  = lane & 15;
  const int hi   = lane >> 4;
  const int w    = tid >> 5;
  const int mw   = w & 3;    // M subtile 0..3
  const int nwv  = w >> 2;   // N wave 0..1

  if (tid < MP)
    smask[tid] = (tid < S_DIM && mask[bn * S_DIM + tid] != 0) ? 1 : 0;

  const unsigned short* xrow = Xb + (long long)bn * MP * KP;
  const int arow = tid >> 2, acq = tid & 3;   // A staging: 16B chunk per thread

  auto stage_a = [&](int ks, int buf) {
    cp16(&AsBuf[buf][arow * ASTRIDE + acq * 4],
         xrow + arow * KP + ks * 32 + acq * 8);
  };
  auto stage_b = [&](int nb, int ks, int buf) {
    const unsigned* src = W1p + ((long long)ks * KP + nb) * 16;
#pragma unroll
    for (int i = 0; i < 6; ++i) {
      int u = tid + i * 256;
      if (u < NTILE1 * 4)                 // 1408 16B chunks, linear copy
        cp16(&BsBuf[buf][u * 4], src + u * 4);
    }
  };

  const int nc0 = nwv * 176 + l15;        // wave's first subtile column

  for (int npass = 0; npass < 6; ++npass) {
    const int nb = npass * NTILE1;
    v8f acc[NSUB1] = {};

    stage_a(0, 0);
    stage_b(nb, 0, 0);

    for (int ks = 0; ks < KSTEPS; ++ks) {
      const int cur = ks & 1;
      cp_wait();
      __syncthreads();
      if (ks + 1 < KSTEPS) {
        stage_a(ks + 1, cur ^ 1);
        stage_b(nb, ks + 1, cur ^ 1);
      }
      const unsigned* As = AsBuf[cur];
      const unsigned* Bs = BsBuf[cur];

      Frag a;                                     // two ds_load_b128
      {
        int m = mw * 16 + l15;
        a.q[0] = *(const uint4*)&As[m * ASTRIDE + hi * 8];
        a.q[1] = *(const uint4*)&As[m * ASTRIDE + hi * 8 + 4];
      }
      // register-double-buffered B fragments: load t+1 before WMMA t
      Frag bf[2];
      bf[0].q[0] = *(const uint4*)&Bs[nc0 * 16 + hi * 8];
      bf[0].q[1] = *(const uint4*)&Bs[nc0 * 16 + hi * 8 + 4];
#pragma unroll
      for (int t = 0; t < NSUB1; ++t) {
        if (t + 1 < NSUB1) {
          int nc = (nc0 + (t + 1) * 16) * 16 + hi * 8;
          bf[(t + 1) & 1].q[0] = *(const uint4*)&Bs[nc];
          bf[(t + 1) & 1].q[1] = *(const uint4*)&Bs[nc + 4];
        }
        acc[t] = __builtin_amdgcn_wmma_f32_16x16x32_bf16(
            false, a.v, false, bf[t & 1].v, (short)0, acc[t], false, false);
      }
    }

    // epilogue: +b1, relu, masked row-reduction -> hbar
#pragma unroll
    for (int t = 0; t < NSUB1; ++t) {
      int ng = nb + nwv * 176 + t * 16 + l15;
      float bias = (ng < K1) ? b1[ng] : 0.f;
      float part = 0.f;
#pragma unroll
      for (int v = 0; v < 8; ++v) {
        int m = mw * 16 + hi * 8 + v;
        float h = fmaxf(acc[t][v] + bias, 0.f);
        part += smask[m] ? h : 0.f;
      }
      part += __shfl_xor(part, 16);
      if (hi == 0)
        atomicAdd(&hbar[(long long)bn * KP + ng], part);
    }
  }
}

// ---------------------------------------------------------------------------
// GEMM2: out[1024,1024] = (hbar[1024,2112] @ W2 + cnt*b2) / 50
// Grid 128 blocks: 16 M-tiles x 8 N-tiles of 64x128.
// ---------------------------------------------------------------------------
__global__ void __launch_bounds__(256)
gemm2_kernel(const float* __restrict__ hbar,
             const unsigned* __restrict__ W2p,
             const float* __restrict__ b2,
             const float* __restrict__ cnt,
             float* __restrict__ out) {
  __shared__ unsigned AsBuf[2][MP * ASTRIDE];   // 2 x 5.0 KB
  __shared__ unsigned BsBuf[2][128 * 16];       // 2 x 8.0 KB
  __shared__ float    scnt[MP];

  const int tid   = threadIdx.x;
  const int mt    = blockIdx.x >> 3;
  const int nt    = blockIdx.x & 7;
  const int mbase = mt * 64, nbase = nt * 128;
  const int lane = tid & 31, l15 = lane & 15, hi = lane >> 4;
  const int w = tid >> 5, mw = w & 3, nwv = w >> 2;

  if (tid < MP) scnt[tid] = cnt[mbase + tid];

  const int arow = tid >> 2, acq = tid & 3;

  auto stage_a = [&](int ks, int buf) {          // f32 -> bf16 convert
    const float4* f4 =
        (const float4*)(hbar + (long long)(mbase + arow) * KP + ks * 32 + acq * 8);
    float4 p0 = f4[0], p1 = f4[1];
    uint4 d;
    d.x = (unsigned)f2bf(p0.x) | ((unsigned)f2bf(p0.y) << 16);
    d.y = (unsigned)f2bf(p0.z) | ((unsigned)f2bf(p0.w) << 16);
    d.z = (unsigned)f2bf(p1.x) | ((unsigned)f2bf(p1.y) << 16);
    d.w = (unsigned)f2bf(p1.z) | ((unsigned)f2bf(p1.w) << 16);
    *(uint4*)&AsBuf[buf][arow * ASTRIDE + acq * 4] = d;
  };
  auto stage_b = [&](int ks, int buf) {
    const unsigned* src = W2p + ((long long)ks * TOK + nbase) * 16;
#pragma unroll
    for (int i = 0; i < 2; ++i) {
      int u = tid + i * 256;                     // 512 16B chunks, linear copy
      cp16(&BsBuf[buf][u * 4], src + u * 4);
    }
  };

  const int nc0 = nwv * 64 + l15;

  v8f acc[NSUB2] = {};
  stage_a(0, 0);
  stage_b(0, 0);

  for (int ks = 0; ks < KSTEPS; ++ks) {
    const int cur = ks & 1;
    cp_wait();
    __syncthreads();
    if (ks + 1 < KSTEPS) {
      stage_a(ks + 1, cur ^ 1);
      stage_b(ks + 1, cur ^ 1);
    }
    const unsigned* As = AsBuf[cur];
    const unsigned* Bs = BsBuf[cur];

    Frag a;
    {
      int m = mw * 16 + l15;
      a.q[0] = *(const uint4*)&As[m * ASTRIDE + hi * 8];
      a.q[1] = *(const uint4*)&As[m * ASTRIDE + hi * 8 + 4];
    }
    Frag bf[2];
    bf[0].q[0] = *(const uint4*)&Bs[nc0 * 16 + hi * 8];
    bf[0].q[1] = *(const uint4*)&Bs[nc0 * 16 + hi * 8 + 4];
#pragma unroll
    for (int t = 0; t < NSUB2; ++t) {
      if (t + 1 < NSUB2) {
        int nc = (nc0 + (t + 1) * 16) * 16 + hi * 8;
        bf[(t + 1) & 1].q[0] = *(const uint4*)&Bs[nc];
        bf[(t + 1) & 1].q[1] = *(const uint4*)&Bs[nc + 4];
      }
      acc[t] = __builtin_amdgcn_wmma_f32_16x16x32_bf16(
          false, a.v, false, bf[t & 1].v, (short)0, acc[t], false, false);
    }
  }

  const float inv_s = 1.0f / 50.0f;
#pragma unroll
  for (int t = 0; t < NSUB2; ++t) {
    int n = nbase + nwv * 64 + t * 16 + l15;
    float bias = b2[n];
#pragma unroll
    for (int v = 0; v < 8; ++v) {
      int m = mw * 16 + hi * 8 + v;
      out[(long long)(mbase + m) * TOK + n] = (acc[t][v] + scnt[m] * bias) * inv_s;
    }
  }
}

// ---------------------------------------------------------------------------
extern "C" void kernel_launch(void* const* d_in, const int* in_sizes, int n_in,
                              void* d_out, int out_size, void* d_ws, size_t ws_size,
                              hipStream_t stream) {
  (void)in_sizes; (void)n_in; (void)out_size; (void)ws_size;
  const float* emb  = (const float*)d_in[0];
  const float* vis  = (const float*)d_in[1];
  const float* bbox = (const float*)d_in[2];
  const float* kp   = (const float*)d_in[3];
  const int*   mask = (const int*)d_in[4];
  const float* W1   = (const float*)d_in[5];
  const float* b1   = (const float*)d_in[6];
  const float* W2   = (const float*)d_in[7];
  const float* b2   = (const float*)d_in[8];
  float* out = (float*)d_out;

  // workspace layout (all 16B aligned)
  unsigned short* Xb = (unsigned short*)d_ws;               // 1024*64*2112 bf16
  unsigned* W1p = (unsigned*)(Xb + (size_t)BN_TOT * MP * KP); // 66*2112*16 u32
  unsigned* W2p = W1p + (size_t)KSTEPS * KP * 16;           // 66*1024*16 u32
  float* hbar = (float*)(W2p + (size_t)KSTEPS * TOK * 16);  // 1024*2112 f32
  float* cnt  = hbar + (size_t)BN_TOT * KP;                 // 1024 f32

  const long long nx = (long long)BN_TOT * MP * KP;
  pack_x_kernel<<<(int)((nx + 255) / 256), 256, 0, stream>>>(emb, vis, bbox, kp, Xb);

  const long long nw1 = (long long)KSTEPS * KP * 16;
  pack_wp_kernel<<<(int)((nw1 + 255) / 256), 256, 0, stream>>>(W1, W1p, K1, K1, KP);

  const long long nw2 = (long long)KSTEPS * TOK * 16;
  pack_wp_kernel<<<(int)((nw2 + 255) / 256), 256, 0, stream>>>(W2, W2p, K1, TOK, TOK);

  const long long nh = (long long)BN_TOT * KP;
  init_kernel<<<(int)((nh + 255) / 256), 256, 0, stream>>>(hbar, cnt, mask);

  gemm1_kernel<<<BN_TOT, 256, 0, stream>>>(Xb, W1p, b1, mask, hbar);
  gemm2_kernel<<<128, 256, 0, stream>>>(hbar, W2p, b2, cnt, out);
}
